// HurdleEncoderV1_41120016892641
// MI455X (gfx1250) — matrix operand order repaired and checked by reading, not compiled, not run on hardware
//
#include <hip/hip_runtime.h>
#include <hip/hip_bf16.h>

// ---------------------------------------------------------------------------
// RGCN encoder for MI455X (gfx1250, wave32).
// Dense math: v_wmma_f32_16x16x32_bf16 (bf16 A/B, fp32 accumulate).
//   - weights pre-converted to bf16 once per launch (L2-resident)
//   - A tiles staged in LDS in fragment-linear order (1x32B read per lane)
// Irregular math: f32 global atomics for segment mean / pooling.
// ---------------------------------------------------------------------------

#define HDIM 128
#define RREL 10
#define LLAY 6
#define GGR  512
#define OUTD 64
#define INDIM 32

typedef __attribute__((ext_vector_type(16))) __bf16 v16bf;
typedef __attribute__((ext_vector_type(8)))  float  v8f;

static __device__ __forceinline__ __bf16 f2bf(float f) {
    unsigned u = __builtin_bit_cast(unsigned, f);
    u += 0x7FFFu + ((u >> 16) & 1u);            // round-to-nearest-even
    unsigned short h = (unsigned short)(u >> 16);
    return __builtin_bit_cast(__bf16, h);
}

// ---------------------------------------------------------------------------
// fp32 -> bf16 bulk conversion (weights; once per launch)
// ---------------------------------------------------------------------------
__global__ void k_cvt_bf16(const float* __restrict__ s, __bf16* __restrict__ d, int n)
{
    const int i = blockIdx.x * blockDim.x + threadIdx.x;
    if (i < n) d[i] = f2bf(s[i]);
}

// ---------------------------------------------------------------------------
// Tiled WMMA GEMM:  C[M,NC] (+)= rowscale(A)[M,KD] * W[KD,NC]   (W in bf16)
//   block: 128 threads (4 waves) -> 16-row x 128-col tile
//   wave w: two 16x16 tiles at cols w*32, w*32+16
//   A tile in LDS, fragment-linear: lane-major, 16 bf16 per lane
//   optional row scale 1/max(cnt,1), bias, accumulate, ReLU
// ---------------------------------------------------------------------------
template<int KD, int NC>
__global__ void __launch_bounds__(128)
k_gemm_wmma(const float* __restrict__ A,
            const __bf16* __restrict__ W,
            const float* __restrict__ bias,
            float* __restrict__ C, int M,
            const float* __restrict__ cnt, int cntStride,
            int accumulate, int doRelu)
{
    __shared__ __align__(32) __bf16 As[16 * 32];   // fragment-order A tile
    __shared__ float inv[16];

    const int tid     = threadIdx.x;
    const int wave    = tid >> 5;
    const int lane    = tid & 31;
    const int rowBase = blockIdx.x * 16;

    if (tid < 16) {
        float s = 1.0f;
        const int rr = rowBase + tid;
        if (cnt != nullptr && rr < M)
            s = 1.0f / fmaxf(cnt[(size_t)rr * cntStride], 1.0f);
        inv[tid] = s;
    }

    const int n0   = wave * 32;
    const bool t0  = (n0      < NC);
    const bool t1v = (n0 + 16 < NC);

    v8f acc0 = {};
    v8f acc1 = {};

    #pragma unroll
    for (int kk = 0; kk < KD; kk += 32) {
        __syncthreads();   // inv ready (iter 0) / prior LDS reads done

        // cooperative A tile load: 256 k-pairs, coalesced float2 along k,
        // stored into LDS already in WMMA A-fragment order.
        #pragma unroll
        for (int p = tid; p < 256; p += 128) {
            const int am = p >> 4;            // row in tile 0..15
            const int k2 = (p & 15) * 2;      // even k 0..30
            const int gr = rowBase + am;
            float2 v = make_float2(0.0f, 0.0f);
            if (gr < M) v = *(const float2*)&A[(size_t)gr * KD + kk + k2];
            const float sc = inv[am];
            // inverse fragment mapping: (am,k2) -> (lane, slot)
            const int ln = am + ((k2 & 8) ? 16 : 0);
            const int vv = ((k2 & 16) >> 2) + ((k2 & 7) >> 1);
            __bf16* dptr = &As[ln * 16 + 2 * vv];
            dptr[0] = f2bf(v.x * sc);
            dptr[1] = f2bf(v.y * sc);
        }
        __syncthreads();

        // A fragment: one aligned 32B LDS read per lane
        const v16bf afrag = *(const v16bf*)&As[lane * 16];

        // B fragments: lane holds column (lane&15), k = 8h+s (s<8), 16+8h+(s-8) (s>=8)
        const __bf16* wrow = W + (size_t)(kk + (lane >> 4) * 8) * NC + (lane & 15);
        if (t0) {
            v16bf b0;
            #pragma unroll
            for (int s = 0; s < 16; ++s)
                b0[s] = wrow[(size_t)((s & 7) + ((s & 8) ? 16 : 0)) * NC + n0];
            acc0 = __builtin_amdgcn_wmma_f32_16x16x32_bf16(
                       false, afrag, false, b0, (short)0, acc0, false, false);
        }
        if (t1v) {
            v16bf b1;
            #pragma unroll
            for (int s = 0; s < 16; ++s)
                b1[s] = wrow[(size_t)((s & 7) + ((s & 8) ? 16 : 0)) * NC + n0 + 16];
            acc1 = __builtin_amdgcn_wmma_f32_16x16x32_bf16(
                       false, afrag, false, b1, (short)0, acc1, false, false);
        }
    }

    // epilogue: lanes 0-15 -> N=lane, M=v ; lanes 16-31 -> N=lane-16, M=v+8
    const int cn = lane & 15;
    const int mo = (lane >> 4) * 8;
    #pragma unroll
    for (int v = 0; v < 8; ++v) {
        const int gm = rowBase + mo + v;
        if (gm >= M) continue;
        if (t0) {
            const size_t idx = (size_t)gm * NC + n0 + cn;
            float val = acc0[v];
            if (bias)        val += bias[n0 + cn];
            if (accumulate)  val += C[idx];
            if (doRelu)      val  = fmaxf(val, 0.0f);
            C[idx] = val;
        }
        if (t1v) {
            const size_t idx = (size_t)gm * NC + n0 + 16 + cn;
            float val = acc1[v];
            if (bias)        val += bias[n0 + 16 + cn];
            if (accumulate)  val += C[idx];
            if (doRelu)      val  = fmaxf(val, 0.0f);
            C[idx] = val;
        }
    }
}

// ---------------------------------------------------------------------------
// Per-(node,relation) in-degree counts (layer-invariant; computed once)
// ---------------------------------------------------------------------------
__global__ void k_count(const int* __restrict__ dst, const int* __restrict__ et,
                        float* __restrict__ cnt, int E)
{
    const int e = blockIdx.x * blockDim.x + threadIdx.x;
    if (e < E) atomicAdd(&cnt[(size_t)dst[e] * RREL + et[e]], 1.0f);
}

// ---------------------------------------------------------------------------
// Scatter-add h[src] into agg[dst] for edges of one relation.
// 16 threads/edge, 8 channels each.
// ---------------------------------------------------------------------------
__global__ void k_scatter(const int* __restrict__ src, const int* __restrict__ dst,
                          const int* __restrict__ et, const float* __restrict__ h,
                          float* __restrict__ agg, int E, int rel)
{
    const long long t = (long long)blockIdx.x * blockDim.x + threadIdx.x;
    const int e = (int)(t >> 4);
    if (e >= E) return;
    if (et[e] != rel) return;
    const int c = ((int)t & 15) * 8;
    const float* hs = h   + (size_t)src[e] * HDIM + c;
    float*       ad = agg + (size_t)dst[e] * HDIM + c;
    #pragma unroll
    for (int i = 0; i < 8; ++i) atomicAdd(ad + i, hs[i]);
}

// ---------------------------------------------------------------------------
// Fused LayerNorm + ReLU + residual:  h = relu(LN(conv)*g + b) + h
// ---------------------------------------------------------------------------
__global__ void __launch_bounds__(128)
k_ln_relu_res(const float* __restrict__ conv, const float* __restrict__ lg,
              const float* __restrict__ lb, float* __restrict__ h)
{
    const int row  = blockIdx.x;
    const int c    = threadIdx.x;
    const int wave = c >> 5;

    __shared__ float ws[4];

    const size_t idx = (size_t)row * HDIM + c;
    const float v    = conv[idx];
    const float res  = h[idx];

    float s = v;
    #pragma unroll
    for (int o = 16; o > 0; o >>= 1) s += __shfl_xor(s, o, 32);
    if ((c & 31) == 0) ws[wave] = s;
    __syncthreads();
    const float mean = (ws[0] + ws[1] + ws[2] + ws[3]) * (1.0f / HDIM);
    __syncthreads();

    const float d  = v - mean;
    float s2 = d * d;
    #pragma unroll
    for (int o = 16; o > 0; o >>= 1) s2 += __shfl_xor(s2, o, 32);
    if ((c & 31) == 0) ws[wave] = s2;
    __syncthreads();
    const float var = (ws[0] + ws[1] + ws[2] + ws[3]) * (1.0f / HDIM);

    float y = d * rsqrtf(var + 1e-5f) * lg[c] + lb[c];
    y = fmaxf(y, 0.0f);
    h[idx] = y + res;
}

// ---------------------------------------------------------------------------
// global_add_pool: g[batch[i]] += h[i]
// ---------------------------------------------------------------------------
__global__ void k_pool(const float* __restrict__ h, const int* __restrict__ batch,
                       float* __restrict__ g, int N)
{
    const long long t = (long long)blockIdx.x * blockDim.x + threadIdx.x;
    const int i = (int)(t >> 7);
    if (i >= N) return;
    const int c = (int)t & 127;
    atomicAdd(&g[(size_t)batch[i] * HDIM + c], h[(size_t)i * HDIM + c]);
}

// ---------------------------------------------------------------------------
extern "C" void kernel_launch(void* const* d_in, const int* in_sizes, int n_in,
                              void* d_out, int out_size, void* d_ws, size_t ws_size,
                              hipStream_t stream)
{
    const float* x      = (const float*)d_in[0];
    const int*   ei     = (const int*)  d_in[1];
    const int*   etype  = (const int*)  d_in[2];
    const int*   batch  = (const int*)  d_in[3];
    const float* W_in   = (const float*)d_in[4];
    const float* b_in   = (const float*)d_in[5];
    const float* W_rel  = (const float*)d_in[6];
    const float* W_root = (const float*)d_in[7];
    const float* b_conv = (const float*)d_in[8];
    const float* ln_g   = (const float*)d_in[9];
    const float* ln_b   = (const float*)d_in[10];
    const float* mW1    = (const float*)d_in[11];
    const float* mb1    = (const float*)d_in[12];
    const float* mW2    = (const float*)d_in[13];
    const float* mb2    = (const float*)d_in[14];
    const float* vW1    = (const float*)d_in[15];
    const float* vb1    = (const float*)d_in[16];
    const float* vW2    = (const float*)d_in[17];
    const float* vb2    = (const float*)d_in[18];
    float* out          = (float*)d_out;

    const int N  = in_sizes[0] / INDIM;   // 50000
    const int E  = in_sizes[2];           // 800000
    const int* src = ei;
    const int* dst = ei + E;

    // -------- workspace carve-up: fp32 region --------
    float* h    = (float*)d_ws;              // N*H   activations
    float* conv = h    + (size_t)N * HDIM;   // N*H   conv accumulator
    float* agg  = conv + (size_t)N * HDIM;   // N*H   per-relation aggregation
    float* cnt  = agg  + (size_t)N * HDIM;   // N*R   in-degree counts
    float* gbuf = cnt  + (size_t)N * RREL;   // G*H   pooled graph features
    float* t1   = gbuf + (size_t)GGR * HDIM; // G*H   head hidden
    float* fend = t1   + (size_t)GGR * HDIM;

    // -------- bf16 weight region (64B aligned) --------
    uintptr_t bp = ((uintptr_t)fend + 63u) & ~(uintptr_t)63u;
    __bf16* wb     = (__bf16*)bp;
    __bf16* bW_in  = wb;                                     // 32*128
    __bf16* bW_root= bW_in   + INDIM * HDIM;                 // 6*128*128
    __bf16* bW_rel = bW_root + LLAY * HDIM * HDIM;           // 60*128*128
    __bf16* bmW1   = bW_rel  + LLAY * RREL * HDIM * HDIM;    // 128*128
    __bf16* bmW2   = bmW1    + HDIM * HDIM;                  // 128*64
    __bf16* bvW1   = bmW2    + HDIM * OUTD;                  // 128*128
    __bf16* bvW2   = bvW1    + HDIM * HDIM;                  // 128*64

    const int gemmGrdN = (N + 15) / 16;
    const int gemmGrdG = (GGR + 15) / 16;

    // -------- one-time weight conversions (cheap; L2-resident afterwards) ---
    {
        const int nIn   = INDIM * HDIM;
        const int nRoot = LLAY * HDIM * HDIM;
        const int nRel  = LLAY * RREL * HDIM * HDIM;
        const int nH    = HDIM * HDIM;
        const int nO    = HDIM * OUTD;
        k_cvt_bf16<<<(nIn   + 255) / 256, 256, 0, stream>>>(W_in,   bW_in,   nIn);
        k_cvt_bf16<<<(nRoot + 255) / 256, 256, 0, stream>>>(W_root, bW_root, nRoot);
        k_cvt_bf16<<<(nRel  + 255) / 256, 256, 0, stream>>>(W_rel,  bW_rel,  nRel);
        k_cvt_bf16<<<(nH    + 255) / 256, 256, 0, stream>>>(mW1,    bmW1,    nH);
        k_cvt_bf16<<<(nO    + 255) / 256, 256, 0, stream>>>(mW2,    bmW2,    nO);
        k_cvt_bf16<<<(nH    + 255) / 256, 256, 0, stream>>>(vW1,    bvW1,    nH);
        k_cvt_bf16<<<(nO    + 255) / 256, 256, 0, stream>>>(vW2,    bvW2,    nO);
    }

    // -------- relation counts (layer-invariant) --------
    hipMemsetAsync(cnt, 0, (size_t)N * RREL * sizeof(float), stream);
    k_count<<<(E + 255) / 256, 256, 0, stream>>>(dst, etype, cnt, E);

    // -------- input projection: h = x @ W_in + b_in --------
    k_gemm_wmma<INDIM, HDIM><<<gemmGrdN, 128, 0, stream>>>(
        x, bW_in, b_in, h, N, nullptr, 0, 0, 0);

    // -------- 6 RGCN layers --------
    for (int l = 0; l < LLAY; ++l) {
        // conv = h @ W_root[l] + b_conv[l]
        k_gemm_wmma<HDIM, HDIM><<<gemmGrdN, 128, 0, stream>>>(
            h, bW_root + (size_t)l * HDIM * HDIM,
            b_conv + (size_t)l * HDIM, conv, N, nullptr, 0, 0, 0);

        for (int r = 0; r < RREL; ++r) {
            hipMemsetAsync(agg, 0, (size_t)N * HDIM * sizeof(float), stream);
            const long long st = (long long)E * 16;
            k_scatter<<<(unsigned)((st + 255) / 256), 256, 0, stream>>>(
                src, dst, etype, h, agg, E, r);
            // conv += (agg / max(cnt_r,1)) @ W_rel[l][r]
            k_gemm_wmma<HDIM, HDIM><<<gemmGrdN, 128, 0, stream>>>(
                agg, bW_rel + ((size_t)l * RREL + r) * HDIM * HDIM,
                nullptr, conv, N, cnt + r, RREL, 1, 0);
        }

        // h = relu(LN(conv)) + h
        k_ln_relu_res<<<N, HDIM, 0, stream>>>(
            conv, ln_g + (size_t)l * HDIM, ln_b + (size_t)l * HDIM, h);
    }

    // -------- global add pool --------
    hipMemsetAsync(gbuf, 0, (size_t)GGR * HDIM * sizeof(float), stream);
    {
        const long long pt = (long long)N * HDIM;
        k_pool<<<(unsigned)((pt + 255) / 256), 256, 0, stream>>>(h, batch, gbuf, N);
    }

    // -------- heads --------
    k_gemm_wmma<HDIM, HDIM><<<gemmGrdG, 128, 0, stream>>>(
        gbuf, bmW1, mb1, t1, GGR, nullptr, 0, 0, 1);
    k_gemm_wmma<HDIM, OUTD><<<gemmGrdG, 128, 0, stream>>>(
        t1, bmW2, mb2, out, GGR, nullptr, 0, 0, 0);
    k_gemm_wmma<HDIM, HDIM><<<gemmGrdG, 128, 0, stream>>>(
        gbuf, bvW1, vb1, t1, GGR, nullptr, 0, 0, 1);
    k_gemm_wmma<HDIM, OUTD><<<gemmGrdG, 128, 0, stream>>>(
        t1, bvW2, vb2, out + (size_t)GGR * OUTD, GGR, nullptr, 0, 0, 0);
}
